// Flow_Attention_67611375173886
// MI455X (gfx1250) — matrix operand order repaired and checked by reading, not compiled, not run on hardware
//
#include <hip/hip_runtime.h>
#include <hip/hip_bf16.h>
#include <math.h>

// Flow-Attention for MI455X (gfx1250, wave32).
// B=8, L=8192, H=8, D=M=64.  Memory-bound: ~1.07 GB total traffic -> ~46us @ 23.3 TB/s.
// WMMA bf16 handles the two 64-wide contractions (kv, qkv); fragments are built
// from LDS with 16B ds_load_b128 chunks (pass5 in ISA table k order, pass4 in contiguous k order).
// Workspace: ~1.1 MB of d_ws (slice-level accumulators + kv).

#define LL   8192
#define HH   8
#define DD   64
#define FEPS 1e-6f

typedef __attribute__((ext_vector_type(16))) __bf16 v16bf;
typedef __attribute__((ext_vector_type(8)))  __bf16 v8bf;
typedef __attribute__((ext_vector_type(8)))  float  v8f;

// ---- workspace layout (float offsets) ----
#define OFF_SUMQ   0                       // [64][64] sum_l sig(q)
#define OFF_SUMK   4096                    // [64][64] sum_l sig(k)
#define OFF_QW     8192                    // [64][64] sum_l sig(q)*sink_in
#define OFF_KW     12288                   // [64][64] sum_l sig(k)*src_out
#define OFF_SUMEXP 16384                   // [64]
#define OFF_KV     16448                   // [64][64][64] K^T (V*exp)
#define N_ACC      (OFF_KV + 262144)       // floats to zero each call (~1.1MB)

__device__ __forceinline__ float sigm(float x) { return 1.0f / (1.0f + __expf(-x)); }

__device__ __forceinline__ v16bf cat16(v8bf lo, v8bf hi) {
    return __builtin_shufflevector(lo, hi, 0, 1, 2, 3, 4, 5, 6, 7,
                                           8, 9, 10, 11, 12, 13, 14, 15);
}

__global__ void __launch_bounds__(256) k_zero(float* ws, int n) {
    int i = blockIdx.x * blockDim.x + threadIdx.x;
    if (i < n) ws[i] = 0.0f;
}

// ---------- Pass 1: sum_l sigmoid(q), sigmoid(k) per (slice, d) ----------
__global__ void __launch_bounds__(256) k_pass1(const float* __restrict__ q,
                                               const float* __restrict__ k,
                                               float* __restrict__ ws) {
    const int s = blockIdx.x;                 // slice = n*H + h
    const int n = s >> 3, h = s & 7;
    const int tid = threadIdx.x, d = tid & 63, row = tid >> 6;
    const int lbase = blockIdx.y * 512;       // 16 chunks of 512 tokens
    float accq = 0.f, acck = 0.f;
    for (int it = 0; it < 128; ++it) {
        int l = lbase + row + it * 4;
        size_t idx = ((size_t)(n * LL + l) * HH + h) * DD + d;
        accq += sigm(q[idx]);
        acck += sigm(k[idx]);
    }
    __shared__ float red[4][64];
    red[row][d] = accq; __syncthreads();
    if (tid < 64) atomicAdd(&ws[OFF_SUMQ + s * 64 + d],
                            red[0][d] + red[1][d] + red[2][d] + red[3][d]);
    __syncthreads();
    red[row][d] = acck; __syncthreads();
    if (tid < 64) atomicAdd(&ws[OFF_SUMK + s * 64 + d],
                            red[0][d] + red[1][d] + red[2][d] + red[3][d]);
}

// ---------- Pass 2: qw = sum sig(q)*sink_in, kw = sum sig(k)*src_out ----------
// one token per wave; butterfly shfl reduce -> no inner barriers
__global__ void __launch_bounds__(256) k_pass2(const float* __restrict__ q,
                                               const float* __restrict__ k,
                                               float* __restrict__ ws) {
    const int s = blockIdx.x;
    const int n = s >> 3, h = s & 7;
    const int tid = threadIdx.x, wave = tid >> 5, lane = tid & 31;
    // per-lane slice sums held in registers (d = lane, lane+32)
    const float smk0 = ws[OFF_SUMK + s * 64 + lane] + FEPS;
    const float smk1 = ws[OFF_SUMK + s * 64 + lane + 32] + FEPS;
    const float smq0 = ws[OFF_SUMQ + s * 64 + lane] + FEPS;
    const float smq1 = ws[OFF_SUMQ + s * 64 + lane + 32] + FEPS;
    const int lbase = blockIdx.y * 512;
    float qw0 = 0.f, qw1 = 0.f, kw0 = 0.f, kw1 = 0.f;
    for (int it = 0; it < 64; ++it) {
        int l = lbase + wave + it * 8;
        size_t idx = ((size_t)(n * LL + l) * HH + h) * DD + lane;
        float sq0 = sigm(q[idx]),      sq1 = sigm(q[idx + 32]);
        float sk0 = sigm(k[idx]),      sk1 = sigm(k[idx + 32]);
        float dq = (sq0 + FEPS) * smk0 + (sq1 + FEPS) * smk1;
        float dk = (sk0 + FEPS) * smq0 + (sk1 + FEPS) * smq1;
#pragma unroll
        for (int o = 1; o < 32; o <<= 1) { dq += __shfl_xor(dq, o, 32);
                                           dk += __shfl_xor(dk, o, 32); }
        float si = 1.0f / dq;   // sink_incoming
        float so = 1.0f / dk;   // source_outgoing
        qw0 += sq0 * si;  qw1 += sq1 * si;
        kw0 += sk0 * so;  kw1 += sk1 * so;
    }
    __shared__ float red[8][128];
    red[wave][lane] = qw0;  red[wave][lane + 32] = qw1;
    red[wave][lane + 64] = kw0;  red[wave][lane + 96] = kw1;
    __syncthreads();
    if (tid < 128) {
        float v = 0.f;
#pragma unroll
        for (int w = 0; w < 8; ++w) v += red[w][tid];
        int off = (tid < 64) ? (OFF_QW + s * 64 + tid) : (OFF_KW + s * 64 + tid - 64);
        atomicAdd(&ws[off], v);
    }
}

// ---------- Pass 4: kv[d][m] = sum_l sig(k)[l][d] * (v[l][m]*exp(cs[l])) ----------
// conserved_source / exp computed in-pass (8-lane reduce); WMMA bf16.
__global__ void __launch_bounds__(256) k_pass4(const float* __restrict__ k_,
                                               const float* __restrict__ v_,
                                               float* __restrict__ ws) {
    const int s = blockIdx.x, split = blockIdx.y;   // 4-way L split
    const int n = s >> 3, h = s & 7;
    const int tid = threadIdx.x, lane = tid & 31, wave = tid >> 5;
    const int half = lane >> 4, li = lane & 15;
    __shared__ alignas(16) __bf16 sklT[64][40];     // [d][l-chunk], 80B rows
    __shared__ alignas(16) __bf16 vnlT[64][40];     // [m][l-chunk]
    const int t = tid >> 3;                         // token 0..31 for staging
    const int d0 = (tid & 7) * 8;
    // per-thread qw (for conserved_source dot), in registers
    float qw[8];
#pragma unroll
    for (int u = 0; u < 8; ++u) qw[u] = ws[OFF_QW + s * 64 + d0 + u] + FEPS;
    const int dt = wave & 3;                        // d-tile (rows of kv)
    const int mt0 = (wave >> 2) * 2;                // two m-tiles per wave
    v8f c0 = {}; v8f c1 = {};
    const int l0 = split * 2048;
    float esum = 0.f;
    for (int step = 0; step < 64; ++step) {
        int lb = l0 + step * 32;
        size_t base = ((size_t)(n * LL + lb + t) * HH + h) * DD + d0;
        float4 ka = *(const float4*)&k_[base];
        float4 kb = *(const float4*)&k_[base + 4];
        float sk[8] = { sigm(ka.x), sigm(ka.y), sigm(ka.z), sigm(ka.w),
                        sigm(kb.x), sigm(kb.y), sigm(kb.z), sigm(kb.w) };
        float pd = 0.f;
#pragma unroll
        for (int u = 0; u < 8; ++u) pd += (sk[u] + FEPS) * qw[u];
        pd += __shfl_xor(pd, 1, 32);
        pd += __shfl_xor(pd, 2, 32);
        pd += __shfl_xor(pd, 4, 32);                // dot over all 64 d
        float e = __expf(fminf(1.0f, fmaxf(-1.0f, pd)));
        if ((tid & 7) == 0) esum += e;
        float4 va = *(const float4*)&v_[base];
        float4 vb = *(const float4*)&v_[base + 4];
        float vv[8] = { va.x, va.y, va.z, va.w, vb.x, vb.y, vb.z, vb.w };
#pragma unroll
        for (int u = 0; u < 8; ++u) {               // transposed staging
            sklT[d0 + u][t] = (__bf16)sk[u];
            vnlT[d0 + u][t] = (__bf16)(vv[u] * e);
        }
        __syncthreads();
        // fragments: two 16B chunks each (ds_load_b128)
        const int ar = dt * 16 + li;
        v16bf a  = cat16(*(const v8bf*)&sklT[ar][16 * half],
                         *(const v8bf*)&sklT[ar][16 * half + 8]);
        const int br0 = (mt0 + 0) * 16 + li, br1 = (mt0 + 1) * 16 + li;
        v16bf b0 = cat16(*(const v8bf*)&vnlT[br0][16 * half],
                         *(const v8bf*)&vnlT[br0][16 * half + 8]);
        v16bf b1 = cat16(*(const v8bf*)&vnlT[br1][16 * half],
                         *(const v8bf*)&vnlT[br1][16 * half + 8]);
        c0 = __builtin_amdgcn_wmma_f32_16x16x32_bf16(false, a, false, b0, (short)0, c0, false, false);
        c1 = __builtin_amdgcn_wmma_f32_16x16x32_bf16(false, a, false, b1, (short)0, c1, false, false);
        __syncthreads();
    }
    if ((tid & 7) == 0) atomicAdd(&ws[OFF_SUMEXP + s], esum);
#pragma unroll
    for (int r = 0; r < 8; ++r) {
        int drow = dt * 16 + r + (half << 3);
        atomicAdd(&ws[OFF_KV + s * 4096 + drow * 64 + (mt0 + 0) * 16 + li], c0[r]);
        atomicAdd(&ws[OFF_KV + s * 4096 + drow * 64 + (mt0 + 1) * 16 + li], c1[r]);
    }
}

// ---------- Pass 5: out = sig(q) @ kv * (alloc * sink_in * 8192/sumexp) ----------
// conserved_sink -> alloc and sink_incoming computed in-pass (2-lane reduce).
__global__ void __launch_bounds__(256) k_pass5(const float* __restrict__ q_,
                                               float* __restrict__ out,
                                               const float* __restrict__ ws) {
    const int s = blockIdx.x, tb = blockIdx.y;      // 32 blocks of 256 tokens
    const int n = s >> 3, h = s & 7;
    const int tid = threadIdx.x, lane = tid & 31, wave = tid >> 5;
    const int half = lane >> 4, li = lane & 15;
    __shared__ alignas(16) __bf16 kvT[64][72];      // [m][d], 144B rows
    __shared__ alignas(16) __bf16 qnl[128][72];     // [token][d]
    __shared__ float allc[128];
    __shared__ float s_kw[64], s_smk[64];
    if (tid < 64) { s_kw[tid]  = ws[OFF_KW   + s * 64 + tid] + FEPS;
                    s_smk[tid] = ws[OFF_SUMK + s * 64 + tid] + FEPS; }
    const float snorm = 8192.0f / ws[OFF_SUMEXP + s];
    // stage kv transposed as bf16
    for (int r = 0; r < 16; ++r) {
        int f = tid + 256 * r;
        kvT[f & 63][f >> 6] = (__bf16)ws[OFF_KV + s * 4096 + f];
    }
    __syncthreads();
    // B fragments (4 m-tiles x 2 k-halves) held in VGPRs, 16B chunk loads
    v16bf bf[4][2];
#pragma unroll
    for (int mt = 0; mt < 4; ++mt)
#pragma unroll
        for (int kh = 0; kh < 2; ++kh) {
            const int br = mt * 16 + li, off = kh * 32 + 8 * half;
            bf[mt][kh] = cat16(*(const v8bf*)&kvT[br][off],
                               *(const v8bf*)&kvT[br][off + 16]);
        }
    const int lblock = tb * 256;
    for (int batch = 0; batch < 2; ++batch) {
        __syncthreads();
        const int lbase = lblock + batch * 128;
        const int t = tid >> 1, d0 = (tid & 1) * 32;
        size_t base = ((size_t)(n * LL + lbase + t) * HH + h) * DD + d0;
        float sq[32];
#pragma unroll
        for (int c4 = 0; c4 < 8; ++c4) {
            float4 qa = *(const float4*)&q_[base + c4 * 4];
            sq[c4 * 4 + 0] = sigm(qa.x); sq[c4 * 4 + 1] = sigm(qa.y);
            sq[c4 * 4 + 2] = sigm(qa.z); sq[c4 * 4 + 3] = sigm(qa.w);
        }
        float pdk = 0.f, pdq = 0.f;
#pragma unroll
        for (int u = 0; u < 32; ++u) {
            pdk += (sq[u] + FEPS) * s_kw[d0 + u];
            pdq += (sq[u] + FEPS) * s_smk[d0 + u];
        }
        pdk += __shfl_xor(pdk, 1, 32);              // full-64 dots
        pdq += __shfl_xor(pdq, 1, 32);
        if ((tid & 1) == 0)
            allc[t] = sigm(pdk) * (1.0f / pdq) * snorm;  // alloc * sink_in * norm
#pragma unroll
        for (int c8 = 0; c8 < 4; ++c8) {            // vector LDS stores
            v8bf pk;
#pragma unroll
            for (int e = 0; e < 8; ++e) pk[e] = (__bf16)sq[c8 * 8 + e];
            *(v8bf*)&qnl[t][d0 + c8 * 8] = pk;
        }
        __syncthreads();
        const int tw = wave * 16;                   // 16 tokens per wave
        const int ar = tw + li;
        v16bf a0 = cat16(*(const v8bf*)&qnl[ar][8 * half],
                         *(const v8bf*)&qnl[ar][16 + 8 * half]);
        v16bf a1 = cat16(*(const v8bf*)&qnl[ar][32 + 8 * half],
                         *(const v8bf*)&qnl[ar][48 + 8 * half]);
#pragma unroll
        for (int mt = 0; mt < 4; ++mt) {
            v8f c = {};
            c = __builtin_amdgcn_wmma_f32_16x16x32_bf16(false, a0, false, bf[mt][0], (short)0, c, false, false);
            c = __builtin_amdgcn_wmma_f32_16x16x32_bf16(false, a1, false, bf[mt][1], (short)0, c, false, false);
#pragma unroll
            for (int r = 0; r < 8; ++r) {
                int trow = tw + r + (half << 3);
                int l = lbase + trow;
                out[((size_t)(n * LL + l) * HH + h) * DD + mt * 16 + li] = c[r] * allc[trow];
            }
        }
    }
}

extern "C" void kernel_launch(void* const* d_in, const int* in_sizes, int n_in,
                              void* d_out, int out_size, void* d_ws, size_t ws_size,
                              hipStream_t stream) {
    const float* q = (const float*)d_in[0];
    const float* k = (const float*)d_in[1];
    const float* v = (const float*)d_in[2];
    float* out = (float*)d_out;
    float* ws  = (float*)d_ws;   // needs ~1.1 MB

    k_zero<<<(N_ACC + 255) / 256, 256, 0, stream>>>(ws, N_ACC);
    dim3 g12(64, 16);
    k_pass1<<<g12, 256, 0, stream>>>(q, k, ws);
    k_pass2<<<g12, 256, 0, stream>>>(q, k, ws);
    k_pass4<<<dim3(64, 4), 256, 0, stream>>>(k, v, ws);
    k_pass5<<<dim3(64, 32), 256, 0, stream>>>(q, out, ws);
}
